// GCN_28398323761562
// MI455X (gfx1250) — matrix operand order repaired and checked by reading, not compiled, not run on hardware
//
#include <hip/hip_runtime.h>
#include <hip/hip_bf16.h>

typedef __attribute__((ext_vector_type(16))) __bf16 v16bf;
typedef __attribute__((ext_vector_type(8)))  __bf16 v8bf;
typedef __attribute__((ext_vector_type(8)))  float  v8f;

__device__ __forceinline__ unsigned short f2bf(float f) {
    // round-to-nearest-even fp32 -> bf16 (ignores NaN payload subtleties)
    unsigned int u = __float_as_uint(f);
    unsigned int r = (u + 0x7FFFu + ((u >> 16) & 1u)) >> 16;
    return (unsigned short)r;
}

// ---------------------------------------------------------------- utilities
__global__ void zero_f32(float* __restrict__ p, int n) {
    int i = blockIdx.x * blockDim.x + threadIdx.x;
    if (i < n) p[i] = 0.0f;
}

__global__ void degree_kernel(const int* __restrict__ src, const int* __restrict__ dst,
                              float* deg_out, float* deg_in, int E) {
    int e = blockIdx.x * blockDim.x + threadIdx.x;
    if (e >= E) return;
    unsafeAtomicAdd(&deg_out[src[e]], 1.0f);
    unsafeAtomicAdd(&deg_in[dst[e]], 1.0f);
}

__global__ void rsqrt_kernel(float* d, int n) {
    int i = blockIdx.x * blockDim.x + threadIdx.x;
    if (i < n) d[i] = rsqrtf(fmaxf(d[i], 1.0f));
}

// W [K x C] row-major (fp32)  ->  Wt [C x K] bf16 (transposed for WMMA-B loads)
__global__ void prep_weight(const float* __restrict__ W, unsigned short* __restrict__ Wt,
                            int K, int C) {
    int t = blockIdx.x * blockDim.x + threadIdx.x;
    if (t >= K * C) return;
    int k = t / C, n = t % C;
    Wt[n * K + k] = f2bf(W[t]);
}

// v = in; v *= rsA[row]?; v += bias[col]?; relu?; v *= rsB[row]?; -> outF / outB
__global__ void ep_kernel(const float* __restrict__ in,
                          const float* __restrict__ rsA,
                          const float* __restrict__ bias,
                          const float* __restrict__ rsB,
                          int relu,
                          float* __restrict__ outF,
                          unsigned short* __restrict__ outB,
                          int M, int C) {
    int t = blockIdx.x * blockDim.x + threadIdx.x;
    if (t >= M * C) return;
    int row = t / C, col = t % C;
    float v = in[t];
    if (rsA)  v *= rsA[row];
    if (bias) v += bias[col];
    if (relu) v = fmaxf(v, 0.0f);
    if (rsB)  v *= rsB[row];
    if (outF) outF[t] = v;
    if (outB) outB[t] = f2bf(v);
}

// agg[dst[e]] += h[src[e]]  (float4-granular gather, native f32 atomics)
__global__ void scatter_kernel(const float* __restrict__ h, const int* __restrict__ src,
                               const int* __restrict__ dst, float* __restrict__ agg,
                               int E, int C) {
    int c4 = C >> 2;
    int t = blockIdx.x * blockDim.x + threadIdx.x;
    if (t >= E * c4) return;
    int e = t / c4, c = (t % c4) << 2;
    const float4 v = *reinterpret_cast<const float4*>(h + (size_t)src[e] * C + c);
    float* p = agg + (size_t)dst[e] * C + c;
    unsafeAtomicAdd(p + 0, v.x);
    unsafeAtomicAdd(p + 1, v.y);
    unsafeAtomicAdd(p + 2, v.z);
    unsafeAtomicAdd(p + 3, v.w);
}

// ---------------------------------------------------------------- WMMA GEMM
// out[M x NT*16] = A[M x 128](bf16) * Bt^T   (Bt stored [NT*16 x 128] bf16)
// epilogue: +bias? relu? *rowscale[row]?  -> fp32 and/or bf16 outputs
template <int NT>
__global__ __launch_bounds__(256) void gemm_bf16(
    const unsigned short* __restrict__ A,
    const unsigned short* __restrict__ Bt,
    float* __restrict__ outF,
    unsigned short* __restrict__ outB,
    const float* __restrict__ bias,
    const float* __restrict__ rowscale,
    int M, int relu) {
    const int wave  = threadIdx.x >> 5;
    const int lane  = threadIdx.x & 31;
    const int strip = blockIdx.x * 8 + wave;      // 16-row strip per wave
    const int nstrips = M >> 4;                   // M is a multiple of 16
    if (strip >= nstrips) return;                 // wave-uniform; EXEC stays all-1s
    const int half = lane >> 4;                   // 0: lanes 0-15, 1: lanes 16-31
    const int lr   = lane & 15;
    const int r0   = strip << 4;
    const unsigned short* arow = A + (size_t)(r0 + lr) * 128;

    v8f acc[NT] = {};
#pragma unroll
    for (int kk = 0; kk < 4; ++kk) {              // K = 4 x 32
        // A 16x32 bf16 fragment: lane<16 holds K {0..15}, lane>=16 holds K {16..31}
        v8bf lo = *reinterpret_cast<const v8bf*>(arow + kk * 32 + half * 16);
        v8bf hi = *reinterpret_cast<const v8bf*>(arow + kk * 32 + half * 16 + 8);
        v16bf a = __builtin_shufflevector(lo, hi, 0, 1, 2, 3, 4, 5, 6, 7,
                                                  8, 9, 10, 11, 12, 13, 14, 15);
#pragma unroll
        for (int nt = 0; nt < NT; ++nt) {
            // B 32x16 bf16 fragment: lane holds col (nt*16+lr), 16 contiguous K values
            v16bf b = *reinterpret_cast<const v16bf*>(
                Bt + (size_t)(nt * 16 + lr) * 128 + kk * 32 + half * 16);
            acc[nt] = __builtin_amdgcn_wmma_f32_16x16x32_bf16(
                false, a, false, b, (short)0, acc[nt], false, false);
        }
    }

    const int C = NT * 16;
#pragma unroll
    for (int nt = 0; nt < NT; ++nt) {
        int col  = nt * 16 + lr;
        float bv = bias ? bias[col] : 0.0f;
#pragma unroll
        for (int i = 0; i < 8; ++i) {             // C/D: VGPR i, row = half*8+i, col = lr
            int row = r0 + half * 8 + i;
            float v = acc[nt][i] + bv;
            if (relu)     v = fmaxf(v, 0.0f);
            if (rowscale) v *= rowscale[row];
            if (outF) outF[(size_t)row * C + col] = v;
            if (outB) outB[(size_t)row * C + col] = f2bf(v);
        }
    }
}

// ---------------------------------------------------------------- launcher
extern "C" void kernel_launch(void* const* d_in, const int* in_sizes, int n_in,
                              void* d_out, int out_size, void* d_ws, size_t ws_size,
                              hipStream_t stream) {
    const float* x    = (const float*)d_in[0];
    const int*   src  = (const int*)d_in[1];
    const int*   dst  = (const int*)d_in[2];
    const float* W0   = (const float*)d_in[3];
    const float* b0   = (const float*)d_in[4];
    const float* fcW  = (const float*)d_in[5];
    const float* fcb  = (const float*)d_in[6];
    const float* fc2W = (const float*)d_in[7];
    const float* fc2b = (const float*)d_in[8];
    const float* W1   = (const float*)d_in[9];
    const float* b1   = (const float*)d_in[10];
    const float* W2   = (const float*)d_in[11];
    const float* b2   = (const float*)d_in[12];

    const int N = in_sizes[0] / 128;
    const int E = in_sizes[1];

    // workspace carve-up (256B aligned slabs)
    auto up = [](size_t v) { return (v + 255) & ~(size_t)255; };
    char* base = (char*)d_ws;
    size_t off = 0;
    float* deg = (float*)(base + off); off += up((size_t)2 * N * 4);   // nsrc=deg, ndst=deg+N
    float* h   = (float*)(base + off); off += up((size_t)N * 128 * 4);
    float* agg = (float*)(base + off); off += up((size_t)N * 128 * 4);
    unsigned short* xb   = (unsigned short*)(base + off); off += up((size_t)N * 128 * 2);
    unsigned short* xb2  = (unsigned short*)(base + off); off += up((size_t)N * 128 * 2);
    unsigned short* W0t  = (unsigned short*)(base + off); off += up(128 * 128 * 2);
    unsigned short* fcWt = (unsigned short*)(base + off); off += up(128 * 128 * 2);
    unsigned short* fc2Wt= (unsigned short*)(base + off); off += up(128 * 128 * 2);
    unsigned short* W1t  = (unsigned short*)(base + off); off += up(128 * 128 * 2);
    unsigned short* W2t  = (unsigned short*)(base + off); off += up(64 * 128 * 2);
    (void)ws_size; (void)n_in; (void)out_size;

    float* nsrc = deg;
    float* ndst = deg + N;

    const int T = 256;
    auto g1 = [&](int n) { return (n + T - 1) / T; };
    const int gridg = ((N >> 4) + 7) / 8;   // 8 waves (strips) per block

    // degree norms
    zero_f32<<<g1(2 * N), T, 0, stream>>>(deg, 2 * N);
    degree_kernel<<<g1(E), T, 0, stream>>>(src, dst, nsrc, ndst, E);
    rsqrt_kernel<<<g1(2 * N), T, 0, stream>>>(deg, 2 * N);

    // weights -> transposed bf16
    prep_weight<<<g1(128 * 128), T, 0, stream>>>(W0,   W0t,   128, 128);
    prep_weight<<<g1(128 * 128), T, 0, stream>>>(fcW,  fcWt,  128, 128);
    prep_weight<<<g1(128 * 128), T, 0, stream>>>(fc2W, fc2Wt, 128, 128);
    prep_weight<<<g1(128 * 128), T, 0, stream>>>(W1,   W1t,   128, 128);
    prep_weight<<<g1(128 * 64),  T, 0, stream>>>(W2,   W2t,   128, 64);

    // ---- layer 0: GraphConv(x,W0,b0) -> relu -> fc -> relu -> fc2 -> relu
    ep_kernel<<<g1(N * 128), T, 0, stream>>>(x, nsrc, nullptr, nullptr, 0,
                                             nullptr, xb, N, 128);          // xb = bf16(x*nsrc)
    gemm_bf16<8><<<gridg, T, 0, stream>>>(xb, W0t, h, nullptr, nullptr, nullptr, N, 0);
    zero_f32<<<g1(N * 128), T, 0, stream>>>(agg, N * 128);
    scatter_kernel<<<g1(E * 32), T, 0, stream>>>(h, src, dst, agg, E, 128);
    ep_kernel<<<g1(N * 128), T, 0, stream>>>(agg, ndst, b0, nullptr, 1,
                                             nullptr, xb, N, 128);          // relu(agg*ndst+b0)
    gemm_bf16<8><<<gridg, T, 0, stream>>>(xb,  fcWt,  nullptr, xb2, fcb,  nullptr, N, 1);
    gemm_bf16<8><<<gridg, T, 0, stream>>>(xb2, fc2Wt, nullptr, xb,  fc2b, nsrc,    N, 1);
    // xb = relu(h@fc2W+fc2b) * nsrc   (prescaled input for layer-1 conv)

    // ---- layer 1: GraphConv -> relu  (and prescale by nsrc for layer 2)
    gemm_bf16<8><<<gridg, T, 0, stream>>>(xb, W1t, h, nullptr, nullptr, nullptr, N, 0);
    zero_f32<<<g1(N * 128), T, 0, stream>>>(agg, N * 128);
    scatter_kernel<<<g1(E * 32), T, 0, stream>>>(h, src, dst, agg, E, 128);
    ep_kernel<<<g1(N * 128), T, 0, stream>>>(agg, ndst, b1, nsrc, 1,
                                             nullptr, xb, N, 128);

    // ---- layer 2: GraphConv (128 -> 64), no activation
    gemm_bf16<4><<<gridg, T, 0, stream>>>(xb, W2t, h, nullptr, nullptr, nullptr, N, 0);
    zero_f32<<<g1(N * 64), T, 0, stream>>>(agg, N * 64);
    scatter_kernel<<<g1(E * 16), T, 0, stream>>>(h, src, dst, agg, E, 64);
    ep_kernel<<<g1(N * 64), T, 0, stream>>>(agg, ndst, b2, nullptr, 0,
                                            (float*)d_out, nullptr, N, 64);
}
